// RNN_MIN_76682346103000
// MI455X (gfx1250) — matrix operand order repaired and checked by reading, not compiled, not run on hardware
//
#include <hip/hip_runtime.h>
#include <hip/hip_bf16.h>

// ---------------------------------------------------------------------------
// LSTM for MI455X (gfx1250, wave32, WMMA).
//
// xg = inputs @ (W_in @ W_ih^T) + (b_in@W_ih^T + b_ih + b_hh)  -- rank-10
// update instead of a 275-GFLOP precompute. Recurrent h @ W_hh^T runs on
// v_wmma_f32_16x16x32_bf16 with W_hh bf16 (8 MB, L2-resident).
//
// Wave tiling: each wave owns 2 m-tiles x (4 gates x 1 h-subtile): 12 b128
// loads per 8 WMMAs; k-loop unrolled x2 for ILP (kills WAR v_nop bubbles).
//
// v3: h history is written as bf16 slices of hs[T,B,H]; hin = slice t-1,
// hout = slice t (no ping-pong, no cross-step buffer hazards). The output
// projection runs ONCE after the scan, off the sequential critical path,
// leaving only the 64 lstm_step launches in the dependent chain. Falls back
// to the interleaved per-step projection if ws_size is too small.
// ---------------------------------------------------------------------------

typedef __attribute__((ext_vector_type(16))) __bf16 v16bf;
typedef __attribute__((ext_vector_type(8)))  __bf16 v8bf;
typedef __attribute__((ext_vector_type(8)))  float  v8f;
typedef __attribute__((ext_vector_type(4)))  float  v4f;

#define T_STEPS 64
#define BATCH   512
#define SDIM    10
#define HDIM    1024
#define GDIM    4096   // 4*HDIM

__device__ __forceinline__ float sigf(float x)  { return 1.0f / (1.0f + __expf(-x)); }
__device__ __forceinline__ float tanhf_(float x){ return 2.0f / (1.0f + __expf(-2.0f * x)) - 1.0f; }

// --- W_eff[s][g] = sum_h W_in[s][h] * W_ih[g][h];  bias_eff[g] = b_in.W_ih[g,:] + b_ih + b_hh
__global__ void prep_weff(const float* __restrict__ W_in, const float* __restrict__ b_in,
                          const float* __restrict__ W_ih, const float* __restrict__ b_ih,
                          const float* __restrict__ b_hh,
                          float* __restrict__ W_eff, float* __restrict__ bias_eff) {
    int g = blockIdx.x * blockDim.x + threadIdx.x;
    if (g >= GDIM) return;
    float acc[SDIM];
    #pragma unroll
    for (int s = 0; s < SDIM; ++s) acc[s] = 0.0f;
    float bacc = b_ih[g] + b_hh[g];
    const float* wrow = W_ih + (size_t)g * HDIM;
    for (int h = 0; h < HDIM; ++h) {
        float w = wrow[h];
        bacc += b_in[h] * w;
        #pragma unroll
        for (int s = 0; s < SDIM; ++s) acc[s] += W_in[s * HDIM + h] * w;
    }
    #pragma unroll
    for (int s = 0; s < SDIM; ++s) W_eff[(size_t)s * GDIM + g] = acc[s];
    bias_eff[g] = bacc;
}

// --- pack W_hh (f32, row-major [4H][H]) -> bf16 row-major
__global__ void pack_whh(const float* __restrict__ W, __bf16* __restrict__ Wb, int n) {
    int i = (blockIdx.x * blockDim.x + threadIdx.x) * 4;
    if (i + 3 < n) {
        v4f v = *(const v4f*)(W + i);
        Wb[i + 0] = (__bf16)v.x;
        Wb[i + 1] = (__bf16)v.y;
        Wb[i + 2] = (__bf16)v.z;
        Wb[i + 3] = (__bf16)v.w;
    }
}

// --- h_{-1} = 0, c0 = 0
__global__ void init_state(float* __restrict__ c, __bf16* __restrict__ h0, int n) {
    int i = blockIdx.x * blockDim.x + threadIdx.x;
    if (i < n) { c[i] = 0.0f; h0[i] = (__bf16)0.0f; }
}

// --- one LSTM timestep: gates = h@W_hh^T (WMMA) + inputs_t@W_eff (rank-10, LDS) + bias
// grid (8 b-tiles, 16 h-tiles), 256 threads = 8 waves. Block tile: 64 b x 64 h x 4 gates.
// wave w: mGrp = w>>2 -> rows [mGrp*32, +32); jSub = w&3 -> cols [jSub*16, +16), gates 0..3.
__global__ __launch_bounds__(256)
void lstm_step(const float* __restrict__ inputs,    // [T,B,S] f32
               const __bf16* __restrict__ Whh,      // [4H,H]  bf16 row-major
               const float* __restrict__ W_eff,     // [S,4H]
               const float* __restrict__ bias_eff,  // [4H]
               const __bf16* __restrict__ hin,      // [B,H] bf16 (t-1)
               __bf16* __restrict__ hout,           // [B,H] bf16 (t)
               float* __restrict__ cstate,          // [B,H] f32
               int t) {
    __shared__ float ldsX[64 * SDIM];        // inputs_t tile  [64 rows][10]
    __shared__ float ldsW[SDIM * 4 * 64];    // W_eff slice    [s][q][64 cols]
    __shared__ float ldsB[4 * 64];           // bias slice     [q][64 cols]

    const int tid  = threadIdx.x;
    const int bBlk = blockIdx.x * 64;
    const int hBlk = blockIdx.y * 64;

    // Stage the rank-10 input contribution operands into LDS.
    for (int idx = tid; idx < 64 * SDIM; idx += 256)
        ldsX[idx] = inputs[((size_t)t * BATCH + bBlk + idx / SDIM) * SDIM + idx % SDIM];
    for (int idx = tid; idx < SDIM * 256; idx += 256) {
        int s = idx >> 8, rem = idx & 255, q = rem >> 6, lc = rem & 63;
        ldsW[idx] = W_eff[(size_t)s * GDIM + q * HDIM + hBlk + lc];
    }
    ldsB[tid] = bias_eff[(tid >> 6) * HDIM + hBlk + (tid & 63)];
    __syncthreads();

    const int lane   = tid & 31;
    const int wave   = tid >> 5;
    const int mGrp   = wave >> 2;   // 0..1 : which 32-row half
    const int jSub   = wave & 3;    // 0..3 : which 16-col h-subtile
    const int nLane  = lane & 15;
    const int hiHalf = lane >> 4;   // lane half selects K sub-block per ISA layout

    // A-matrix row per m-subtile for this lane (row-major bf16 h).
    size_t hinRow[2];
    #pragma unroll
    for (int m = 0; m < 2; ++m)
        hinRow[m] = (size_t)(bBlk + (2 * mGrp + m) * 16 + nLane) * HDIM;

    // B-matrix (W_hh) row per gate for this lane: lane = column N = h column within the 16-col subtile.
    size_t bRow[4];
    #pragma unroll
    for (int q = 0; q < 4; ++q)
        bRow[q] = (size_t)(q * HDIM + hBlk + jSub * 16 + nLane) * HDIM;

    v8f zero = {};
    v8f acc[2][4];
    #pragma unroll
    for (int m = 0; m < 2; ++m)
        #pragma unroll
        for (int q = 0; q < 4; ++q) acc[m][q] = zero;

    #pragma unroll 2
    for (int kb = 0; kb < HDIM; kb += 32) {
        // A fragments (16x32 bf16): contiguous 16B loads, K split by lane half.
        const int aO1 = kb + (hiHalf ? 8 : 0);
        const int aO2 = kb + (hiHalf ? 24 : 16);
        v16bf A[2];
        #pragma unroll
        for (int m = 0; m < 2; ++m) {
            v8bf alo = *(const v8bf*)(hin + hinRow[m] + aO1);
            v8bf ahi = *(const v8bf*)(hin + hinRow[m] + aO2);
            #pragma unroll
            for (int e = 0; e < 8; ++e) { A[m][e] = alo[e]; A[m][e + 8] = ahi[e]; }
        }
        // B fragments (32x16 bf16), one per gate; each feeds 2 WMMAs.
        const int kB = kb + (hiHalf ? 8 : 0);
        #pragma unroll
        for (int q = 0; q < 4; ++q) {
            v8bf blo = *(const v8bf*)(Whh + bRow[q] + kB);
            v8bf bhi = *(const v8bf*)(Whh + bRow[q] + kB + 16);
            v16bf B;
            #pragma unroll
            for (int e = 0; e < 8; ++e) { B[e] = blo[e]; B[e + 8] = bhi[e]; }
            #pragma unroll
            for (int m = 0; m < 2; ++m)
                acc[m][q] = __builtin_amdgcn_wmma_f32_16x16x32_bf16(
                    false, A[m], false, B, (short)0, acc[m][q], false, false);
        }
    }

    // Epilogue: C/D layout -> lane owns col N=lane&15, rows M=v+8*(lane>=16).
    // Column is fixed per wave, so per-column operands load from LDS once.
    const int lc   = jSub * 16 + nLane;
    const int hcol = hBlk + lc;
    float be[4], xw[4][SDIM];
    #pragma unroll
    for (int q = 0; q < 4; ++q) {
        be[q] = ldsB[q * 64 + lc];
        #pragma unroll
        for (int s = 0; s < SDIM; ++s) xw[q][s] = ldsW[(s * 4 + q) * 64 + lc];
    }
    #pragma unroll
    for (int m = 0; m < 2; ++m) {
        #pragma unroll
        for (int v = 0; v < 8; ++v) {
            int M    = v + 8 * hiHalf;
            int lrow = (2 * mGrp + m) * 16 + M;
            float gv[4];
            #pragma unroll
            for (int q = 0; q < 4; ++q) gv[q] = acc[m][q][v] + be[q];
            #pragma unroll
            for (int s = 0; s < SDIM; ++s) {
                float xs = ldsX[lrow * SDIM + s];
                #pragma unroll
                for (int q = 0; q < 4; ++q) gv[q] += xs * xw[q][s];
            }
            float ig = sigf(gv[0]);
            float fg = sigf(gv[1]);
            float gg = tanhf_(gv[2]);
            float og = sigf(gv[3]);
            size_t cidx = (size_t)(bBlk + lrow) * HDIM + hcol;
            float cn = fg * cstate[cidx] + ig * gg;
            cstate[cidx] = cn;
            hout[cidx] = (__bf16)(og * tanhf_(cn));
        }
    }
}

// --- out[b, tBase+tl, :] = h_tl @ W_out + b_out over nT timestep slices of hs.
// Full overwrite of its slice -> deterministic, no init needed.
__global__ void out_proj(const __bf16* __restrict__ hs,   // [nT,B,H] bf16
                         const float* __restrict__ Wout,  // [H,S]
                         const float* __restrict__ bout,  // [S]
                         float* __restrict__ out,         // [B,T,S]
                         int tBase, int nT) {
    int idx = blockIdx.x * blockDim.x + threadIdx.x;
    if (idx >= nT * BATCH * SDIM) return;
    int s   = idx % SDIM;
    int rem = idx / SDIM;
    int b   = rem % BATCH;
    int tl  = rem / BATCH;
    const __bf16* hr = hs + ((size_t)tl * BATCH + b) * HDIM;
    float acc = bout[s];
    for (int h = 0; h < HDIM; h += 8) {
        v8bf hv = *(const v8bf*)(hr + h);
        #pragma unroll
        for (int e = 0; e < 8; ++e) acc += (float)hv[e] * Wout[(h + e) * SDIM + s];
    }
    out[((size_t)b * T_STEPS + tBase + tl) * SDIM + s] = acc;
}

extern "C" void kernel_launch(void* const* d_in, const int* in_sizes, int n_in,
                              void* d_out, int out_size, void* d_ws, size_t ws_size,
                              hipStream_t stream) {
    const float* inputs = (const float*)d_in[0];
    const float* W_in   = (const float*)d_in[1];
    const float* b_in   = (const float*)d_in[2];
    const float* W_ih   = (const float*)d_in[3];
    const float* W_hh   = (const float*)d_in[4];
    const float* b_ih   = (const float*)d_in[5];
    const float* b_hh   = (const float*)d_in[6];
    const float* W_out  = (const float*)d_in[7];
    const float* b_out  = (const float*)d_in[8];
    float* out = (float*)d_out;

    char* ws = (char*)d_ws;
    size_t off = 0;
    auto alloc = [&](size_t bytes) -> void* {
        void* p = ws + off;
        off = (off + bytes + 255) & ~(size_t)255;
        return p;
    };
    const size_t BH = (size_t)BATCH * HDIM;
    __bf16* WhhB = (__bf16*)alloc((size_t)GDIM * HDIM * 2);   // 8 MB
    float*  Weff = (float*) alloc((size_t)SDIM * GDIM * 4);   // 160 KB
    float*  beff = (float*) alloc((size_t)GDIM * 4);          // 16 KB
    float*  cst  = (float*) alloc(BH * 4);                    // 2 MB
    __bf16* h0   = (__bf16*)alloc(BH * 2);                    // 1 MB (zeros, h_{-1})

    const size_t hsBytes = (size_t)T_STEPS * BH * 2;          // 64 MB history
    const bool fused = (ws_size > off + 512) && (ws_size - off - 512 >= hsBytes);

    prep_weff<<<GDIM / 256, 256, 0, stream>>>(W_in, b_in, W_ih, b_ih, b_hh, Weff, beff);
    pack_whh<<<(GDIM * HDIM / 4 + 255) / 256, 256, 0, stream>>>(W_hh, WhhB, GDIM * HDIM);
    init_state<<<((int)BH + 255) / 256, 256, 0, stream>>>(cst, h0, (int)BH);

    if (fused) {
        // h history: slice t of hs is h_t. Pure chain of 64 lstm_step launches,
        // then one bulk output projection (1280 blocks, fully parallel).
        __bf16* hs = (__bf16*)alloc(hsBytes);
        for (int t = 0; t < T_STEPS; ++t) {
            const __bf16* hin = (t == 0) ? h0 : hs + (size_t)(t - 1) * BH;
            lstm_step<<<dim3(8, 16), 256, 0, stream>>>(inputs, WhhB, Weff, beff,
                                                       hin, hs + (size_t)t * BH, cst, t);
        }
        out_proj<<<(T_STEPS * BATCH * SDIM + 255) / 256, 256, 0, stream>>>(
            hs, W_out, b_out, out, 0, T_STEPS);
    } else {
        // Small-workspace fallback: ping-pong h, per-step projection.
        __bf16* hA = (__bf16*)alloc(BH * 2);
        __bf16* hB = (__bf16*)alloc(BH * 2);
        // reuse h0 as the t=0 input; hA starts as the t=0 output.
        for (int t = 0; t < T_STEPS; ++t) {
            const __bf16* hin = (t == 0) ? h0 : ((t & 1) ? hA : hB);
            __bf16*      hout = (t & 1) ? hB : hA;
            lstm_step<<<dim3(8, 16), 256, 0, stream>>>(inputs, WhhB, Weff, beff,
                                                       hin, hout, cst, t);
            out_proj<<<(BATCH * SDIM + 255) / 256, 256, 0, stream>>>(
                hout, W_out, b_out, out, t, 1);
        }
    }
}
